// DecoderBlock_77867757076909
// MI455X (gfx1250) — matrix operand order repaired and checked by reading, not compiled, not run on hardware
//
#include <hip/hip_runtime.h>
#include <hip/hip_bf16.h>
#include <stdint.h>
#include <math.h>

typedef __bf16 bf16;
typedef __bf16 v16bf __attribute__((ext_vector_type(16)));
typedef float  v8f   __attribute__((ext_vector_type(8)));
typedef uint32_t u32x4 __attribute__((ext_vector_type(4)));
typedef int32_t  i32x4 __attribute__((ext_vector_type(4)));
typedef int32_t  i32x8 __attribute__((ext_vector_type(8)));

#define WMMA_BF16(a, b, c) \
  __builtin_amdgcn_wmma_f32_16x16x32_bf16(false, (a), false, (b), (short)0, (c), false, false)

#if __has_builtin(__builtin_amdgcn_tensor_load_to_lds) && \
    __has_builtin(__builtin_amdgcn_s_wait_tensorcnt)
#define HAVE_TDM 1
#else
#define HAVE_TDM 0
#endif

// ---------------------------------------------------------------------------
// Problem constants (B=2, T=2048, E=1024, H=16, D=64, DF=4096)
// ---------------------------------------------------------------------------
#define BB   2
#define TT   2048
#define EE   1024
#define HH   16
#define DD   64
#define DFF  4096
#define MM   (BB * TT)        // 4096 rows

// ---------------------------------------------------------------------------
// Fragment loads from LDS, matching CDNA5 WMMA VGPR layouts (ISA 7.12.2)
// ---------------------------------------------------------------------------
__device__ inline v16bf load_a_frag(const bf16* base, int ld, int m0, int k0) {
  int lane = threadIdx.x & 31;
  const uint32_t* row =
      reinterpret_cast<const uint32_t*>(base + (size_t)(m0 + (lane & 15)) * ld);
  int kh = ((lane >> 4) << 3) + k0;
  union { uint32_t u[8]; v16bf v; } t;
#pragma unroll
  for (int i = 0; i < 4; ++i) {
    t.u[i]     = row[(kh + 2 * i) >> 1];
    t.u[i + 4] = row[(kh + 16 + 2 * i) >> 1];
  }
  return t.v;
}

__device__ inline v16bf load_b_frag(const bf16* base, int ld, int k0, int n0) {
  int lane = threadIdx.x & 31;
  const bf16* col = base + (size_t)k0 * ld + n0 + (lane & 15);
  int kh = (lane >> 4) << 3;
  v16bf v;
#pragma unroll
  for (int i = 0; i < 8; ++i) {
    v[i]     = col[(size_t)(kh + i) * ld];
    v[i + 8] = col[(size_t)(kh + 16 + i) * ld];
  }
  return v;
}

// ---------------------------------------------------------------------------
// TDM: 2D bf16 tile load into LDS with row padding (D# per ISA 8.3/8.4).
//   tile0 = contiguous (dim0) tile length in elements, tile1 = #rows,
//   stride0 = tensor row stride in elements,
//   pad_int: 0=2,1=4,2=8,3=16,4=32,5=64... dwords between pads,
//   pad_amt: dwords-1 of pad inserted.
// ---------------------------------------------------------------------------
#if HAVE_TDM
__device__ inline void tdm_load_2d_bf16(uint32_t lds_off, const bf16* gptr,
                                        uint32_t td0, uint32_t td1,
                                        uint32_t tile0, uint32_t tile1,
                                        uint64_t stride0,
                                        uint32_t pad_int, uint32_t pad_amt) {
  uint64_t ga = (uint64_t)(uintptr_t)gptr;
  u32x4 g0;
  g0[0] = 1u;                                         // count=1 (valid user D#)
  g0[1] = lds_off;                                    // lds_addr [63:32]
  g0[2] = (uint32_t)ga;                               // global_addr lo
  g0[3] = (uint32_t)((ga >> 32) & 0x1FFFFFFu) | (2u << 30);  // addr hi | type=2
  uint32_t flags = (1u << 16)                         // data_size = 1 (2 bytes)
                 | (1u << 20)                         // pad_enable
                 | (pad_int << 22) | (pad_amt << 25); // LDS row padding
  i32x8 g1;
  g1[0] = (int)flags;
  g1[1] = (int)((td0 & 0xFFFFu) << 16);               // tensor_dim0[15:0] @ bit48
  g1[2] = (int)((td0 >> 16) | ((td1 & 0xFFFFu) << 16));
  g1[3] = (int)(((td1 >> 16) & 0xFFFFu) | (tile0 << 16));  // tile_dim0
  g1[4] = (int)(tile1 & 0xFFFFu);                     // tile_dim1; tile_dim2=0
  g1[5] = (int)(uint32_t)stride0;                     // tensor_dim0_stride lo
  g1[6] = (int)(uint32_t)((stride0 >> 32) & 0xFFFFu); // stride hi; dim1_stride=0
  g1[7] = 0;
  i32x4 gz = {0, 0, 0, 0};
#if defined(__clang_major__) && (__clang_major__ >= 23)
  i32x8 gz8 = {0, 0, 0, 0, 0, 0, 0, 0};
  __builtin_amdgcn_tensor_load_to_lds(g0, g1, gz, gz, gz8, 0);
#else
  __builtin_amdgcn_tensor_load_to_lds(g0, g1, gz, gz, 0);
#endif
}
#endif

// ---------------------------------------------------------------------------
// Conversion / packing kernels
// ---------------------------------------------------------------------------
__global__ void cvt_f32_bf16(const float* __restrict__ in, bf16* __restrict__ out, int n) {
  int i = blockIdx.x * blockDim.x + threadIdx.x;
  if (i < n) out[i] = (bf16)in[i];
}

__global__ void pack_qkv_w(const float* __restrict__ Wq, const float* __restrict__ Wk,
                           const float* __restrict__ Wv, bf16* __restrict__ Wqkv) {
  int i = blockIdx.x * blockDim.x + threadIdx.x;   // over H*E*D = 1048576
  if (i >= HH * EE * DD) return;
  int d = i & 63;
  int e = (i >> 6) & 1023;
  int h = i >> 16;
  int col = h * DD + d;
  bf16* row = Wqkv + (size_t)e * (3 * EE);
  row[col]          = (bf16)Wq[i];
  row[EE + col]     = (bf16)Wk[i];
  row[2 * EE + col] = (bf16)Wv[i];
}

// ---------------------------------------------------------------------------
// Tiled WMMA GEMM: C = act(A[M,K] @ B[K,N] + bias) (+resid)
// Block tile 128x128, K-step 32. 8 waves in 4(M) x 2(N); wave tile 32x64.
// TDM double-buffered LDS staging when available.
// ---------------------------------------------------------------------------
#define GBM   128
#define GBN   128
#define GBK   32
#define LDA_S 40    // 16 dwords + 4 pad dwords  (TDM pad_int=3, pad_amt=3)
#define LDB_S 136   // 64 dwords + 4 pad dwords  (TDM pad_int=5, pad_amt=3)

template <bool BIAS, bool RES, bool RELU, bool OF, bool OB>
__global__ void __launch_bounds__(256) gemm_bf16(
    const bf16* __restrict__ A, const bf16* __restrict__ Bm,
    int Mdim, int Ndim, int Kdim,
    const float* __restrict__ bias, const float* __restrict__ resid,
    float* __restrict__ outf, bf16* __restrict__ outb) {
  __shared__ alignas(16) bf16 As[2][GBM * LDA_S];
  __shared__ alignas(16) bf16 Bs[2][GBK * LDB_S];

  const int tid  = threadIdx.x;
  const int lane = tid & 31;
  const int w    = tid >> 5;
  const int wm   = w & 3;
  const int wn   = w >> 2;
  const int m0   = blockIdx.y * GBM;
  const int n0   = blockIdx.x * GBN;

  v8f acc[2][4];
#pragma unroll
  for (int i = 0; i < 2; ++i)
#pragma unroll
    for (int j = 0; j < 4; ++j)
      acc[i][j] = v8f{0.f, 0.f, 0.f, 0.f, 0.f, 0.f, 0.f, 0.f};

  const int nk = Kdim / GBK;

#if HAVE_TDM
  // Prologue: wave 0 issues tile 0 via the Tensor Data Mover.
  if (w == 0) {
    tdm_load_2d_bf16((uint32_t)(uintptr_t)&As[0][0], A + (size_t)m0 * Kdim,
                     (uint32_t)Kdim, (uint32_t)Mdim, GBK, GBM, (uint64_t)Kdim, 3, 3);
    tdm_load_2d_bf16((uint32_t)(uintptr_t)&Bs[0][0], Bm + n0,
                     (uint32_t)Ndim, (uint32_t)Kdim, GBN, GBK, (uint64_t)Ndim, 5, 3);
  }
#endif

  for (int kt = 0; kt < nk; ++kt) {
    const int buf = kt & 1;
#if HAVE_TDM
    __syncthreads();   // all readers of buf^1 (iter kt-1) are done
    if (w == 0) {
      if (kt + 1 < nk) {
        int k1 = (kt + 1) * GBK;
        tdm_load_2d_bf16((uint32_t)(uintptr_t)&As[buf ^ 1][0],
                         A + (size_t)m0 * Kdim + k1,
                         (uint32_t)Kdim, (uint32_t)Mdim, GBK, GBM, (uint64_t)Kdim, 3, 3);
        tdm_load_2d_bf16((uint32_t)(uintptr_t)&Bs[buf ^ 1][0],
                         Bm + (size_t)k1 * Ndim + n0,
                         (uint32_t)Ndim, (uint32_t)Kdim, GBN, GBK, (uint64_t)Ndim, 5, 3);
        __builtin_amdgcn_s_wait_tensorcnt(2);  // tile kt complete, kt+1 in flight
      } else {
        __builtin_amdgcn_s_wait_tensorcnt(0);
      }
    }
    __syncthreads();   // tile kt visible to all waves
#else
    const int k0 = kt * GBK;
#pragma unroll
    for (int it = 0; it < 2; ++it) {
      int u = tid + it * 256;
      int r = u >> 2, c = (u & 3) << 3;
      *reinterpret_cast<uint4*>(&As[buf][r * LDA_S + c]) =
          *reinterpret_cast<const uint4*>(A + (size_t)(m0 + r) * Kdim + k0 + c);
    }
#pragma unroll
    for (int it = 0; it < 2; ++it) {
      int u = tid + it * 256;
      int r = u >> 4, c = (u & 15) << 3;
      *reinterpret_cast<uint4*>(&Bs[buf][r * LDB_S + c]) =
          *reinterpret_cast<const uint4*>(Bm + (size_t)(k0 + r) * Ndim + n0 + c);
    }
    __syncthreads();
#endif

    v16bf af[2], bfv[4];
#pragma unroll
    for (int i = 0; i < 2; ++i) af[i] = load_a_frag(As[buf], LDA_S, wm * 32 + i * 16, 0);
#pragma unroll
    for (int j = 0; j < 4; ++j) bfv[j] = load_b_frag(Bs[buf], LDB_S, 0, wn * 64 + j * 16);

#pragma unroll
    for (int i = 0; i < 2; ++i)
#pragma unroll
      for (int j = 0; j < 4; ++j)
        acc[i][j] = WMMA_BF16(af[i], bfv[j], acc[i][j]);

#if !HAVE_TDM
    __syncthreads();
#endif
  }

  // Straight-line epilogue (flags are compile-time).
  const int rbase = m0 + wm * 32 + ((lane >> 4) << 3);
  const int cbase = n0 + wn * 64 + (lane & 15);
#pragma unroll
  for (int i = 0; i < 2; ++i)
#pragma unroll
    for (int j = 0; j < 4; ++j) {
      int col = cbase + j * 16;
      float bb = BIAS ? bias[col] : 0.0f;
#pragma unroll
      for (int g = 0; g < 8; ++g) {
        int row = rbase + i * 16 + g;
        size_t idx = (size_t)row * Ndim + col;
        float v = acc[i][j][g] + bb;
        if (RELU) v = v > 0.f ? v : 0.f;
        if (RES) v += resid[idx];
        if (OF) outf[idx] = v;
        if (OB) outb[idx] = (bf16)v;
      }
    }
}

// ---------------------------------------------------------------------------
// Flash attention: one block = 64 query rows of one (b,h); 4 waves x 16 rows.
// ---------------------------------------------------------------------------
__global__ void __launch_bounds__(128) attn_kernel(const bf16* __restrict__ qkv,
                                                   bf16* __restrict__ outb) {
  __shared__ alignas(16) bf16  Qs[64 * 72];
  __shared__ alignas(16) bf16  Kts[64 * 72];   // K transposed: [d][key]
  __shared__ alignas(16) bf16  Vs[64 * 72];    // [key][d]
  __shared__ alignas(16) float Ss[64 * 68];
  __shared__ alignas(16) bf16  Ps[64 * 72];
  __shared__ float mrow[64], lrow[64], arow[64];

  const int tid  = threadIdx.x;
  const int lane = tid & 31;
  const int w    = tid >> 5;
  const int qt   = blockIdx.x;
  const int bh   = blockIdx.y;
  const int b    = bh >> 4, h = bh & 15;
  const int q0   = qt * 64;
  const float scale = 0.03125f;   // E^-0.5 (reference scales by n_embed)

  const size_t rs = 3 * EE;
  const bf16* qb = qkv + ((size_t)b * TT + q0) * rs + h * DD;
  const bf16* kc = qkv + ((size_t)b * TT) * rs + EE + h * DD;
  const bf16* vc = qkv + ((size_t)b * TT) * rs + 2 * EE + h * DD;

#pragma unroll
  for (int it = 0; it < 4; ++it) {
    int u = tid + it * 128;
    int r = u >> 3, c = (u & 7) << 3;
    *reinterpret_cast<uint4*>(&Qs[r * 72 + c]) =
        *reinterpret_cast<const uint4*>(qb + (size_t)r * rs + c);
  }
  if (tid < 64) { mrow[tid] = -INFINITY; lrow[tid] = 0.f; }
  __syncthreads();

  v16bf aQ0 = load_a_frag(Qs, 72, w * 16, 0);
  v16bf aQ1 = load_a_frag(Qs, 72, w * 16, 32);

  v8f acc[4];
#pragma unroll
  for (int n = 0; n < 4; ++n) acc[n] = v8f{0.f, 0.f, 0.f, 0.f, 0.f, 0.f, 0.f, 0.f};

  const int r0 = w * 16 + ((lane >> 4) << 3);

  for (int kt = 0; kt <= qt; ++kt) {
    const bf16* kb = kc + (size_t)kt * 64 * rs;
    const bf16* vb = vc + (size_t)kt * 64 * rs;
    if (kt + 1 <= qt) {  // prefetch next key/value tile (global_prefetch_b8)
      __builtin_prefetch(kb + 64 * rs + (size_t)(tid & 63) * rs, 0, 1);
      __builtin_prefetch(vb + 64 * rs + (size_t)(tid & 63) * rs, 0, 1);
    }
#pragma unroll
    for (int it = 0; it < 4; ++it) {
      int u = tid + it * 128;
      int r = u >> 3, c = (u & 7) << 3;
      *reinterpret_cast<uint4*>(&Vs[r * 72 + c]) =
          *reinterpret_cast<const uint4*>(vb + (size_t)r * rs + c);
      union { uint4 q; bf16 e[8]; } tk;
      tk.q = *reinterpret_cast<const uint4*>(kb + (size_t)r * rs + c);
#pragma unroll
      for (int j = 0; j < 8; ++j) Kts[(c + j) * 72 + r] = tk.e[j];
    }
    __syncthreads();

    // S = Q @ K^T
#pragma unroll
    for (int n = 0; n < 4; ++n) {
      v8f s = v8f{0.f, 0.f, 0.f, 0.f, 0.f, 0.f, 0.f, 0.f};
      s = WMMA_BF16(aQ0, load_b_frag(Kts, 72, 0, n * 16), s);
      s = WMMA_BF16(aQ1, load_b_frag(Kts, 72, 32, n * 16), s);
      int cc = n * 16 + (lane & 15);
#pragma unroll
      for (int g = 0; g < 8; ++g) Ss[(r0 + g) * 68 + cc] = s[g];
    }
    __syncthreads();

    // Online softmax, one thread per query row
    if (tid < 64) {
      int r = tid;
      int nvalid = (q0 + r) - kt * 64 + 1;
      if (nvalid > 64) nvalid = 64;
      float mt = -INFINITY;
      for (int c = 0; c < nvalid; ++c) {
        float sv = Ss[r * 68 + c] * scale;
        Ss[r * 68 + c] = sv;
        mt = fmaxf(mt, sv);
      }
      float mold = mrow[r];
      float mnew = fmaxf(mold, mt);
      float alpha = __expf(mold - mnew);
      float psum = 0.f;
      for (int c = 0; c < 64; ++c) {
        float p = (c < nvalid) ? __expf(Ss[r * 68 + c] - mnew) : 0.f;
        Ps[r * 72 + c] = (bf16)p;
        psum += p;
      }
      lrow[r] = lrow[r] * alpha + psum;
      mrow[r] = mnew;
      arow[r] = alpha;
    }
    __syncthreads();

    // Rescale running O and accumulate P @ V
    float al[8];
#pragma unroll
    for (int g = 0; g < 8; ++g) al[g] = arow[r0 + g];
    v16bf aP0 = load_a_frag(Ps, 72, w * 16, 0);
    v16bf aP1 = load_a_frag(Ps, 72, w * 16, 32);
#pragma unroll
    for (int n = 0; n < 4; ++n) {
#pragma unroll
      for (int g = 0; g < 8; ++g) acc[n][g] *= al[g];
      acc[n] = WMMA_BF16(aP0, load_b_frag(Vs, 72, 0, n * 16), acc[n]);
      acc[n] = WMMA_BF16(aP1, load_b_frag(Vs, 72, 32, n * 16), acc[n]);
    }
    __syncthreads();
  }

  float li[8];
#pragma unroll
  for (int g = 0; g < 8; ++g) li[g] = 1.f / lrow[r0 + g];
#pragma unroll
  for (int n = 0; n < 4; ++n) {
    int col = h * DD + n * 16 + (lane & 15);
#pragma unroll
    for (int g = 0; g < 8; ++g) {
      int row = q0 + r0 + g;
      outb[((size_t)b * TT + row) * EE + col] = (bf16)(acc[n][g] * li[g]);
    }
  }
}

// ---------------------------------------------------------------------------
// Host-side orchestration
// ---------------------------------------------------------------------------
extern "C" void kernel_launch(void* const* d_in, const int* in_sizes, int n_in,
                              void* d_out, int out_size, void* d_ws, size_t ws_size,
                              hipStream_t stream) {
  const float* x  = (const float*)d_in[0];
  const float* Wq = (const float*)d_in[1];
  const float* Wk = (const float*)d_in[2];
  const float* Wv = (const float*)d_in[3];
  const float* Wo = (const float*)d_in[4];
  const float* bo = (const float*)d_in[5];
  const float* W1 = (const float*)d_in[6];
  const float* b1 = (const float*)d_in[7];
  const float* W2 = (const float*)d_in[8];
  const float* b2 = (const float*)d_in[9];

  uint8_t* ws = (uint8_t*)d_ws;
  size_t off = 0;
  auto alloc = [&](size_t bytes) -> void* {
    void* p = ws + off;
    off += (bytes + 255) & ~(size_t)255;
    return p;
  };

  bf16* xb    = (bf16*)alloc((size_t)MM * EE * 2);
  bf16* Wqkvb = (bf16*)alloc((size_t)EE * 3 * EE * 2);
  bf16* qkvb  = (bf16*)alloc((size_t)MM * 3 * EE * 2);
  bf16* attnb = (bf16*)alloc((size_t)MM * EE * 2);
  bf16* Wob   = (bf16*)alloc((size_t)EE * EE * 2);
  float* x1   = (float*)alloc((size_t)MM * EE * 4);
  bf16* x1b   = (bf16*)alloc((size_t)MM * EE * 2);
  bf16* W1b   = (bf16*)alloc((size_t)EE * DFF * 2);
  bf16* hb    = (bf16*)alloc((size_t)MM * DFF * 2);
  bf16* W2b   = (bf16*)alloc((size_t)DFF * EE * 2);

  const int ct = 256;
  { int n = MM * EE;  cvt_f32_bf16<<<(n + ct - 1) / ct, ct, 0, stream>>>(x, xb, n); }
  { int n = EE * EE;  cvt_f32_bf16<<<(n + ct - 1) / ct, ct, 0, stream>>>(Wo, Wob, n); }
  { int n = EE * DFF; cvt_f32_bf16<<<(n + ct - 1) / ct, ct, 0, stream>>>(W1, W1b, n); }
  { int n = DFF * EE; cvt_f32_bf16<<<(n + ct - 1) / ct, ct, 0, stream>>>(W2, W2b, n); }
  { int n = HH * EE * DD;
    pack_qkv_w<<<(n + ct - 1) / ct, ct, 0, stream>>>(Wq, Wk, Wv, Wqkvb); }

  // 1) QKV: [4096,1024] @ [1024,3072] -> qkvb (bf16)
  gemm_bf16<false, false, false, false, true>
      <<<dim3(3 * EE / GBN, MM / GBM), 256, 0, stream>>>(
          xb, Wqkvb, MM, 3 * EE, EE, nullptr, nullptr, nullptr, qkvb);

  // 2) Attention (flash, causal) -> attnb (bf16, concat heads)
  attn_kernel<<<dim3(TT / 64, BB * HH), 128, 0, stream>>>(qkvb, attnb);

  // 3) Output projection + bo + residual(x) -> x1 (fp32) and x1b (bf16)
  gemm_bf16<true, true, false, true, true>
      <<<dim3(EE / GBN, MM / GBM), 256, 0, stream>>>(
          attnb, Wob, MM, EE, EE, bo, x, x1, x1b);

  // 4) FFN up: relu(x1 @ W1 + b1) -> hb (bf16)
  gemm_bf16<true, false, true, false, true>
      <<<dim3(DFF / GBN, MM / GBM), 256, 0, stream>>>(
          x1b, W1b, MM, DFF, EE, b1, nullptr, nullptr, hb);

  // 5) FFN down: x1 + (h @ W2 + b2) -> d_out (fp32)
  gemm_bf16<true, true, false, true, false>
      <<<dim3(EE / GBN, MM / GBM), 256, 0, stream>>>(
          hb, W2b, MM, EE, DFF, b2, x1, (float*)d_out, nullptr);
}
